// MetricLearningLoss_55087250538752
// MI455X (gfx1250) — compile-verified, matrix operand order checked by reading, not run on hardware
//
#include <hip/hip_runtime.h>
#include <math.h>

typedef float  v2f   __attribute__((ext_vector_type(2)));
typedef float  v8f   __attribute__((ext_vector_type(8)));
typedef __bf16 bf16_t;
typedef bf16_t v4bf  __attribute__((ext_vector_type(4)));
typedef bf16_t v8bf  __attribute__((ext_vector_type(8)));
typedef bf16_t v16bf __attribute__((ext_vector_type(16)));

#define K_ROWS 4096
#define DIM    512
#define NB     64   // number of 64-row blocks (4096/64)
#define BT     64   // block tile edge

// loss constants (k=4096, sigma=0.2, omega=1.0); constexpr -> inline literals
constexpr float kC_COEF = 2047.0f;                // k/2 - 1
constexpr float kAS     = 1.0f / 327.68f;         // 1/(2k sigma^2)
constexpr float kLS     = 5.792037522411089f;     // ln(2k sigma^2) = ln(327.68)
constexpr float kAO     = 1.0f / 8192.0f;         // 1/(2k omega^2)
constexpr float kLO     = 9.010913347279288f;     // ln(8192)

// ---------------------------------------------------------------------------
// Kernel 1: per-row squared norms (fp32, exact). One wave32 per row.
// ---------------------------------------------------------------------------
__global__ void __launch_bounds__(256) rownorm_kernel(const float* __restrict__ X,
                                                      float* __restrict__ sq) {
    int wid  = threadIdx.x >> 5;
    int lane = threadIdx.x & 31;
    int row  = blockIdx.x * 8 + wid;
    const float* p = X + (size_t)row * DIM;
    float s = 0.0f;
    #pragma unroll
    for (int t = 0; t < DIM / 32; ++t) {
        float v = p[lane + t * 32];
        s += v * v;
    }
    #pragma unroll
    for (int off = 16; off > 0; off >>= 1)
        s += __shfl_down(s, off, 32);
    if (lane == 0) sq[row] = s;
}

// ---------------------------------------------------------------------------
// Kernel 1b: split fp32 -> bf16 hi + bf16 lo planes (x ~= hi + lo).
// ---------------------------------------------------------------------------
__global__ void __launch_bounds__(256) split_kernel(const float* __restrict__ X,
                                                    bf16_t* __restrict__ hi,
                                                    bf16_t* __restrict__ lo) {
    int idx = (blockIdx.x * 256 + threadIdx.x) * 4;
    float4 x = *(const float4*)(X + idx);
    v4bf h, l;
    h[0] = (bf16_t)x.x; h[1] = (bf16_t)x.y; h[2] = (bf16_t)x.z; h[3] = (bf16_t)x.w;
    l[0] = (bf16_t)(x.x - (float)h[0]);
    l[1] = (bf16_t)(x.y - (float)h[1]);
    l[2] = (bf16_t)(x.z - (float)h[2]);
    l[3] = (bf16_t)(x.w - (float)h[3]);
    *(v4bf*)(hi + idx) = h;
    *(v4bf*)(lo + idx) = l;
}

// ---------------------------------------------------------------------------
// Epilogue for one 16x16 fp32 accumulator tile: lane l holds col gn+(l&15);
// VGPR v holds row gm + v + 8*(l>>4).
// ---------------------------------------------------------------------------
__device__ __forceinline__ float tile_loss(v8f c, int gm, int gn, int lane,
                                           const int* __restrict__ lab,
                                           const float* __restrict__ sq) {
    int lhi  = lane >> 4;
    int col  = gn + (lane & 15);
    float sqc = sq[col];
    int   lc  = lab[col];
    float s = 0.0f;
    #pragma unroll
    for (int v = 0; v < 8; ++v) {
        int row = gm + v + 8 * lhi;
        if (row < col) {  // upper-triangular pairs i<j only
            float d2 = sq[row] + sqc - 2.0f * c[v];
            float lg = logf(d2);
            float val = (lab[row] == lc)
                      ? (-kC_COEF * (lg - kLS) + 0.5f * kAS * d2)
                      : ( kC_COEF * (lg - kLO) - 0.5f * kAO * d2);
            s += val;
        }
    }
    return s;
}

__device__ __forceinline__ void block_reduce_store(float acc, int wid, int lane,
                                                   float* __restrict__ partials,
                                                   int pslot) {
    #pragma unroll
    for (int off = 16; off > 0; off >>= 1)
        acc += __shfl_down(acc, off, 32);
    __shared__ float wsum[4];
    if (lane == 0) wsum[wid] = acc;
    __syncthreads();
    if (threadIdx.x == 0)
        partials[pslot] = wsum[0] + wsum[1] + wsum[2] + wsum[3];
}

// ===========================================================================
// PATH A: bf16 split GEMM (3 products) via V_WMMA_F32_16X16X32_BF16.
// A frag (16x32 bf16): lane l = row base+(l&15); two 16B chunks at
// K = k + 8*(l>>4) and K = k + 16 + 8*(l>>4).  B = X^T -> identical pattern.
// ===========================================================================
__device__ __forceinline__ v16bf load_frag_bf(const bf16_t* __restrict__ B,
                                              int row_base, int k, int lane) {
    int r   = row_base + (lane & 15);
    int off = k + ((lane >> 4) << 3);
    const bf16_t* p = B + (size_t)r * DIM + off;
    v8bf c0 = *(const v8bf*)(p);        // K = off .. off+7   (16B aligned)
    v8bf c1 = *(const v8bf*)(p + 16);   // K = off+16 .. +23
    return __builtin_shufflevector(c0, c1, 0, 1, 2, 3, 4, 5, 6, 7,
                                            8, 9, 10, 11, 12, 13, 14, 15);
}

#define WMMA_BF16(A, B, C) \
    __builtin_amdgcn_wmma_f32_16x16x32_bf16(false, (A), false, (B), (short)0, (C), false, false)

__global__ void __launch_bounds__(128) gram_loss_bf16_kernel(
        const bf16_t* __restrict__ Xh, const bf16_t* __restrict__ Xl,
        const int* __restrict__ lab, const float* __restrict__ sq,
        float* __restrict__ partials) {
    int bj = blockIdx.x, bi = blockIdx.y;
    int pslot = bi * NB + bj;
    if (bi > bj) { if (threadIdx.x == 0) partials[pslot] = 0.0f; return; }

    int wid  = threadIdx.x >> 5;
    int lane = threadIdx.x & 31;
    int gm0  = bi * BT + (wid >> 1) * 32;
    int gn0  = bj * BT + (wid & 1) * 32;

    v8f c00 = {}, c01 = {}, c10 = {}, c11 = {};

    #pragma unroll 2
    for (int k = 0; k < DIM; k += 32) {
        v16bf ah0 = load_frag_bf(Xh, gm0,      k, lane);
        v16bf ah1 = load_frag_bf(Xh, gm0 + 16, k, lane);
        v16bf bh0 = load_frag_bf(Xh, gn0,      k, lane);
        v16bf bh1 = load_frag_bf(Xh, gn0 + 16, k, lane);
        v16bf al0 = load_frag_bf(Xl, gm0,      k, lane);
        v16bf al1 = load_frag_bf(Xl, gm0 + 16, k, lane);
        v16bf bl0 = load_frag_bf(Xl, gn0,      k, lane);
        v16bf bl1 = load_frag_bf(Xl, gn0 + 16, k, lane);

        // cross terms first, dominant hi*hi last
        c00 = WMMA_BF16(ah0, bl0, c00);  c00 = WMMA_BF16(al0, bh0, c00);
        c01 = WMMA_BF16(ah0, bl1, c01);  c01 = WMMA_BF16(al0, bh1, c01);
        c10 = WMMA_BF16(ah1, bl0, c10);  c10 = WMMA_BF16(al1, bh0, c10);
        c11 = WMMA_BF16(ah1, bl1, c11);  c11 = WMMA_BF16(al1, bh1, c11);
        c00 = WMMA_BF16(ah0, bh0, c00);
        c01 = WMMA_BF16(ah0, bh1, c01);
        c10 = WMMA_BF16(ah1, bh0, c10);
        c11 = WMMA_BF16(ah1, bh1, c11);
    }

    float acc = tile_loss(c00, gm0,      gn0,      lane, lab, sq)
              + tile_loss(c01, gm0,      gn0 + 16, lane, lab, sq)
              + tile_loss(c10, gm0 + 16, gn0,      lane, lab, sq)
              + tile_loss(c11, gm0 + 16, gn0 + 16, lane, lab, sq);
    block_reduce_store(acc, wid, lane, partials, pslot);
}

// ===========================================================================
// PATH B (fallback, small workspace): fp32 V_WMMA_F32_16X16X4_F32.
// ===========================================================================
__device__ __forceinline__ v2f load_frag_f32(const float* __restrict__ X,
                                             int row_base, int k, int lane) {
    int r  = row_base + (lane & 15);
    int kk = k + ((lane >> 4) << 1);
    return *(const v2f*)(X + (size_t)r * DIM + kk);
}

__global__ void __launch_bounds__(128) gram_loss_f32_kernel(
        const float* __restrict__ X, const int* __restrict__ lab,
        const float* __restrict__ sq, float* __restrict__ partials) {
    int bj = blockIdx.x, bi = blockIdx.y;
    int pslot = bi * NB + bj;
    if (bi > bj) { if (threadIdx.x == 0) partials[pslot] = 0.0f; return; }

    int wid  = threadIdx.x >> 5;
    int lane = threadIdx.x & 31;
    int gm0  = bi * BT + (wid >> 1) * 32;
    int gn0  = bj * BT + (wid & 1) * 32;

    v8f c00 = {}, c01 = {}, c10 = {}, c11 = {};
    #pragma unroll 4
    for (int k = 0; k < DIM; k += 4) {
        v2f a0 = load_frag_f32(X, gm0,      k, lane);
        v2f a1 = load_frag_f32(X, gm0 + 16, k, lane);
        v2f b0 = load_frag_f32(X, gn0,      k, lane);
        v2f b1 = load_frag_f32(X, gn0 + 16, k, lane);
        c00 = __builtin_amdgcn_wmma_f32_16x16x4_f32(false, a0, false, b0, (short)0, c00, false, false);
        c01 = __builtin_amdgcn_wmma_f32_16x16x4_f32(false, a0, false, b1, (short)0, c01, false, false);
        c10 = __builtin_amdgcn_wmma_f32_16x16x4_f32(false, a1, false, b0, (short)0, c10, false, false);
        c11 = __builtin_amdgcn_wmma_f32_16x16x4_f32(false, a1, false, b1, (short)0, c11, false, false);
    }

    float acc = tile_loss(c00, gm0,      gn0,      lane, lab, sq)
              + tile_loss(c01, gm0,      gn0 + 16, lane, lab, sq)
              + tile_loss(c10, gm0 + 16, gn0,      lane, lab, sq)
              + tile_loss(c11, gm0 + 16, gn0 + 16, lane, lab, sq);
    block_reduce_store(acc, wid, lane, partials, pslot);
}

// ---------------------------------------------------------------------------
// Kernel 3: deterministic final reduction (double accumulation), scale 1/(2k).
// ---------------------------------------------------------------------------
__global__ void __launch_bounds__(256) reduce_kernel(const float* __restrict__ partials,
                                                     float* __restrict__ out) {
    __shared__ double sm[256];
    double s = 0.0;
    for (int i = threadIdx.x; i < NB * NB; i += 256)
        s += (double)partials[i];
    sm[threadIdx.x] = s;
    __syncthreads();
    for (int off = 128; off > 0; off >>= 1) {
        if (threadIdx.x < (unsigned)off) sm[threadIdx.x] += sm[threadIdx.x + off];
        __syncthreads();
    }
    if (threadIdx.x == 0)
        out[0] = (float)(sm[0] / (2.0 * (double)K_ROWS));
}

// ---------------------------------------------------------------------------
extern "C" void kernel_launch(void* const* d_in, const int* in_sizes, int n_in,
                              void* d_out, int out_size, void* d_ws, size_t ws_size,
                              hipStream_t stream) {
    const float* X   = (const float*)d_in[0];   // outputs [4096, 512] f32
    const int*   lab = (const int*)d_in[1];     // labels  [4096] int

    const size_t n_elem   = (size_t)K_ROWS * DIM;            // 2 Mi elements
    float*  sq       = (float*)d_ws;                         // 4096 f32
    float*  partials = sq + K_ROWS;                          // 4096 f32
    bf16_t* hi       = (bf16_t*)(partials + NB * NB);        // 4 MB
    bf16_t* lo       = hi + n_elem;                          // 4 MB
    const size_t need_split = (size_t)(K_ROWS + NB * NB) * 4 + 2 * n_elem * 2;

    rownorm_kernel<<<K_ROWS / 8, 256, 0, stream>>>(X, sq);

    if (ws_size >= need_split) {
        split_kernel<<<(int)(n_elem / 4 / 256), 256, 0, stream>>>(X, hi, lo);
        gram_loss_bf16_kernel<<<dim3(NB, NB), 128, 0, stream>>>(hi, lo, lab, sq, partials);
    } else {
        gram_loss_f32_kernel<<<dim3(NB, NB), 128, 0, stream>>>(X, lab, sq, partials);
    }
    reduce_kernel<<<1, 256, 0, stream>>>(partials, (float*)d_out);
}
